// RecurrentGConvLSTM_40278203302521
// MI455X (gfx1250) — matrix operand order repaired and checked by reading, not compiled, not run on hardware
//
#include <hip/hip_runtime.h>

// ---------------------------------------------------------------------------
// Fused GConvLSTM (K=1) cell for MI455X / gfx1250.
//  - GEMMs via V_WMMA_F32_16X16X32_BF16 (bf16 A/B, f32 accumulate)
//  - one wave32 per 16-node tile, weights pre-swizzled into LDS
//  - gate math on VALU (v_exp/v_rcp), head reduction via lane shuffles
// ---------------------------------------------------------------------------

typedef __attribute__((ext_vector_type(16))) __bf16 v16bf;
typedef __attribute__((ext_vector_type(8)))  float  v8f;

#define N_NODES 500000
#define NFEAT   16
#define HID     32
#define NTILES  (N_NODES / 16)      // 31250, exact
#define WAVES_PER_BLOCK 8
#define BLOCK   (WAVES_PER_BLOCK * 32)

__device__ __forceinline__ float fsig(float x) {
    // sigmoid(x) = 1 / (1 + e^-x)   -> v_exp_f32 + v_rcp_f32
    float e = __expf(-x);
    return __builtin_amdgcn_rcpf(1.0f + e);
}
__device__ __forceinline__ float ftanh(float x) {
    // tanh(x) = (1 - e^-2x) / (1 + e^-2x)
    float e = __expf(-2.0f * x);
    return (1.0f - e) * __builtin_amdgcn_rcpf(1.0f + e);
}

__global__ __launch_bounds__(BLOCK)
void gconvlstm_fused_kernel(
    const float* __restrict__ x,      // [N,16]
    const float* __restrict__ h,      // [N,32]
    const float* __restrict__ c,      // [N,32]
    const float* __restrict__ W_x,    // [4,16,32]
    const float* __restrict__ b_x,    // [4,32]
    const float* __restrict__ W_h,    // [4,32,32]
    const float* __restrict__ b_h,    // [4,32]
    const float* __restrict__ w_c,    // [3,32]
    const float* __restrict__ b_g,    // [4,32]
    const float* __restrict__ W_lin,  // [32]
    const float* __restrict__ b_lin,  // [1]
    float* __restrict__ out,          // [N]
    float* __restrict__ h_out,        // [N,32]
    float* __restrict__ c_out)        // [N,32]
{
    // ---- LDS: B fragments pre-swizzled to the WMMA 32x16 bf16 per-lane layout
    //      lane L: column n = L&15, half hh = L>>4; element j<8 holds K = hh*8 + j, j>=8 holds K = 16 + hh*8 + (j-8)
    __shared__ __align__(32) __bf16 sBx[8][32][16];  // [g*2+t][lane][j]
    __shared__ __align__(32) __bf16 sBh[8][32][16];
    __shared__ float sBias[4][32];                   // b_x + b_h + b_g folded
    __shared__ float sWc[3][32];
    __shared__ float sWlin[32];
    __shared__ float sBlin;

    const int tid = threadIdx.x;

    for (int idx = tid; idx < 8 * 32 * 16; idx += BLOCK) {
        const int f  = idx >> 9;          // fragment 0..7
        const int l  = (idx >> 4) & 31;   // lane slot
        const int j  = idx & 15;          // element within lane
        const int g  = f >> 1, t = f & 1;
        const int nn = l & 15, hh = l >> 4;
        const int K  = (j < 8) ? (hh * 8 + j) : (16 + hh * 8 + (j - 8));
        const int col = nn + 16 * t;
        const float vx = (K < NFEAT) ? W_x[(g * NFEAT + K) * HID + col] : 0.0f;
        const float vh = W_h[(g * HID + K) * HID + col];
        sBx[f][l][j] = (__bf16)vx;
        sBh[f][l][j] = (__bf16)vh;
    }
    for (int idx = tid; idx < 4 * 32; idx += BLOCK)
        sBias[idx >> 5][idx & 31] = b_x[idx] + b_h[idx] + b_g[idx];
    for (int idx = tid; idx < 3 * 32; idx += BLOCK)
        sWc[idx / 32][idx % 32] = w_c[idx];
    if (tid < 32)  sWlin[tid] = W_lin[tid];
    if (tid == 0)  sBlin = b_lin[0];
    __syncthreads();

    const int wave = tid >> 5;
    const int lane = tid & 31;
    const int tile = blockIdx.x * WAVES_PER_BLOCK + wave;
    if (tile >= NTILES) return;

    const int tb = tile * 16;
    const int n  = lane & 15;   // column within 16-wide tile / node row for A
    const int hi = lane >> 4;   // half-wave select

    // ---- A fragments: 16x32 bf16, M=n on lanes, K split low/high half-wave
    //      lane(hi=0): elem j<8 -> K=j,     j>=8 -> K=16+(j-8)
    //      lane(hi=1): elem j<8 -> K=8+j,   j>=8 -> K=24+(j-8)
    const int nodeA = tb + n;
    v16bf ax, ah;
    {   // x tile, K padded 16..31 with zero (NFEAT == 16)
        const float4* xp = (const float4*)(x + nodeA * NFEAT + hi * 8);
        const float4 a0 = xp[0], a1 = xp[1];
        ax[0] = (__bf16)a0.x; ax[1] = (__bf16)a0.y; ax[2] = (__bf16)a0.z; ax[3] = (__bf16)a0.w;
        ax[4] = (__bf16)a1.x; ax[5] = (__bf16)a1.y; ax[6] = (__bf16)a1.z; ax[7] = (__bf16)a1.w;
#pragma unroll
        for (int j = 8; j < 16; ++j) ax[j] = (__bf16)0.0f;
    }
    {   // h tile, full K=32
        const float4* hp0 = (const float4*)(h + nodeA * HID + hi * 8);
        const float4* hp1 = (const float4*)(h + nodeA * HID + 16 + hi * 8);
        const float4 a0 = hp0[0], a1 = hp0[1], a2 = hp1[0], a3 = hp1[1];
        ah[0]  = (__bf16)a0.x; ah[1]  = (__bf16)a0.y; ah[2]  = (__bf16)a0.z; ah[3]  = (__bf16)a0.w;
        ah[4]  = (__bf16)a1.x; ah[5]  = (__bf16)a1.y; ah[6]  = (__bf16)a1.z; ah[7]  = (__bf16)a1.w;
        ah[8]  = (__bf16)a2.x; ah[9]  = (__bf16)a2.y; ah[10] = (__bf16)a2.z; ah[11] = (__bf16)a2.w;
        ah[12] = (__bf16)a3.x; ah[13] = (__bf16)a3.y; ah[14] = (__bf16)a3.z; ah[15] = (__bf16)a3.w;
    }

    // ---- gate pre-activations: s[g] = x@Wx[g] + h@Wh[g] + bias[g]
    v8f acc[4][2];
#pragma unroll
    for (int g = 0; g < 4; ++g) {
#pragma unroll
        for (int t = 0; t < 2; ++t) {
            const v16bf bx = *(const v16bf*)&sBx[g * 2 + t][lane][0];
            const v16bf bh = *(const v16bf*)&sBh[g * 2 + t][lane][0];
            v8f a = {};
            a = __builtin_amdgcn_wmma_f32_16x16x32_bf16(false, ax, false, bx,
                                                        (short)0, a, false, false);
            a = __builtin_amdgcn_wmma_f32_16x16x32_bf16(false, ah, false, bh,
                                                        (short)0, a, false, false);
            const float bias = sBias[g][n + 16 * t];
#pragma unroll
            for (int r = 0; r < 8; ++r) a[r] += bias;
            acc[g][t] = a;
        }
    }

    // ---- LSTM gate math directly on the C/D layout:
    //      VGPR r, half hi -> node tb + r + 8*hi; lane n, tile t -> column n+16t
    float pr[8];
#pragma unroll
    for (int r = 0; r < 8; ++r) pr[r] = 0.0f;

#pragma unroll
    for (int t = 0; t < 2; ++t) {
        const int col = n + 16 * t;
        const float wc0 = sWc[0][col], wc1 = sWc[1][col], wc2 = sWc[2][col];
        const float wl  = sWlin[col];
#pragma unroll
        for (int r = 0; r < 8; ++r) {
            const int node = tb + r + 8 * hi;
            const int idx  = node * HID + col;
            const float cv = c[idx];
            const float s0 = acc[0][t][r];
            const float s1 = acc[1][t][r];
            const float s2 = acc[2][t][r];
            const float s3 = acc[3][t][r];
            const float ig = fsig(s0 + wc0 * cv);
            const float fg = fsig(s1 + wc1 * cv);
            const float tg = ftanh(s2);
            const float cn = fg * cv + ig * tg;
            const float og = fsig(s3 + wc2 * cn);
            const float h0 = og * ftanh(cn);
            c_out[idx] = cn;
            h_out[idx] = h0;
            pr[r] += fmaxf(h0, 0.0f) * wl;      // relu(h0) @ W_lin partial
        }
    }

    // ---- 32->1 head: reduce the 16 columns held in each half-wave
    const float bl = sBlin;
#pragma unroll
    for (int r = 0; r < 8; ++r) {
        float v = pr[r];
        v += __shfl_xor(v, 8, 32);
        v += __shfl_xor(v, 4, 32);
        v += __shfl_xor(v, 2, 32);
        v += __shfl_xor(v, 1, 32);
        if (n == 0) out[tb + r + 8 * hi] = v + bl;   // lanes 0 and 16 write
    }
}

extern "C" void kernel_launch(void* const* d_in, const int* in_sizes, int n_in,
                              void* d_out, int out_size, void* d_ws, size_t ws_size,
                              hipStream_t stream) {
    (void)in_sizes; (void)n_in; (void)out_size; (void)d_ws; (void)ws_size;
    const float* x     = (const float*)d_in[0];
    // d_in[1] = edge_index, d_in[2] = edge_weight: mathematically unused (K=1 Cheb)
    const float* h     = (const float*)d_in[3];
    const float* c     = (const float*)d_in[4];
    const float* W_x   = (const float*)d_in[5];
    const float* b_x   = (const float*)d_in[6];
    const float* W_h   = (const float*)d_in[7];
    const float* b_h   = (const float*)d_in[8];
    const float* w_c   = (const float*)d_in[9];
    const float* b_g   = (const float*)d_in[10];
    const float* W_lin = (const float*)d_in[11];
    const float* b_lin = (const float*)d_in[12];

    float* out   = (float*)d_out;                       // [N]
    float* h_out = out + N_NODES;                       // [N,32]
    float* c_out = h_out + (size_t)N_NODES * HID;       // [N,32]

    const int blocks = (NTILES + WAVES_PER_BLOCK - 1) / WAVES_PER_BLOCK;
    gconvlstm_fused_kernel<<<blocks, BLOCK, 0, stream>>>(
        x, h, c, W_x, b_x, W_h, b_h, w_c, b_g, W_lin, b_lin, out, h_out, c_out);
}
